// AMAG_40097814675815
// MI455X (gfx1250) — matrix operand order repaired and checked by reading, not compiled, not run on hardware
//
#include <hip/hip_runtime.h>
#include <hip/hip_bf16.h>
#include <math.h>

typedef __attribute__((ext_vector_type(16))) __bf16 v16bf;
typedef __attribute__((ext_vector_type(8)))  float  v8f;

union FragU { uint4 u[2]; v16bf v; };

static constexpr int B_ = 16, T_ = 20, C_ = 128, F_ = 9, D_ = 64, DFF_ = 256;
static constexpr float EPS_ = 1e-5f;
static constexpr int TD_ = T_ * D_;              // 1280
static constexpr int BC_ = B_ * C_;              // 2048
static constexpr int ROWS_ = B_ * C_ * T_;       // 40960 (fc-mix rows)

#define WMMA_BF16(A, Bf, Cc) \
    __builtin_amdgcn_wmma_f32_16x16x32_bf16(false, (A), false, (Bf), (short)0, (Cc), false, false)

__device__ __forceinline__ v16bf frag_from2(uint4 a, uint4 b) {
    FragU f; f.u[0] = a; f.u[1] = b; return f.v;
}
// A fragment (16x32 tile, row m per lane; lanes<16 K={0..7,16..23}, lanes>=16 K={8..15,24..31})
__device__ __forceinline__ v16bf load_A_frag(const __bf16* base, int stride, int mbase,
                                             int k0, int lane) {
    int mm = lane & 15; bool hi = lane >= 16;
    const __bf16* r = base + (size_t)(mbase + mm) * stride + k0 + (hi ? 8 : 0);
    return frag_from2(*(const uint4*)r, *(const uint4*)(r + 16));
}
// B fragment from a transposed-stored matrix Bt[n][k] (lane holds column n; lanes<16 K={0..7,16..23}, lanes>=16 K={8..15,24..31})
__device__ __forceinline__ v16bf load_Bt_frag(const __bf16* base, int stride, int n0,
                                              int k0, int lane) {
    int nn = lane & 15; bool hi = lane >= 16;
    const __bf16* r = base + (size_t)(n0 + nn) * stride + k0 + (hi ? 8 : 0);
    return frag_from2(*(const uint4*)r, *(const uint4*)(r + 16));
}
// pre-swizzled fragment blob (512 bf16 per fragment, 16 per lane)
__device__ __forceinline__ v16bf g_frag(const __bf16* base, int frag, int lane) {
    const uint4* p = (const uint4*)(base + ((size_t)frag * 32 + lane) * 16);
    return frag_from2(p[0], p[1]);
}
__device__ __forceinline__ int apat(bool hi, int e) {
    return hi ? ((e < 8) ? 8 + e : 16 + e) : ((e < 8) ? e : 8 + e);
}
__device__ __forceinline__ float pe_val(int t, int d) {
    int j = d >> 1;
    float div = expf(-(logf(10000.f) / (float)D_) * (float)(2 * j));
    float ang = (float)t * div;
    return (d & 1) ? cosf(ang) : sinf(ang);
}

// ============ weight swizzle: W[K][N] row-major -> B-fragment blobs (frag = nt*KT+kt)
__global__ void swizzle_w(const float* __restrict__ W, __bf16* __restrict__ out,
                          int K, int N) {
    int idx = blockIdx.x * blockDim.x + threadIdx.x;
    if (idx >= K * N) return;
    int e = idx & 15, lane = (idx >> 4) & 31, frag = idx >> 9;
    int KT = K >> 5;
    int kt = frag % KT, nt = frag / KT;
    int n = nt * 16 + (lane & 15);
    int k = kt * 32 + apat(lane >= 16, e);
    out[idx] = (__bf16)W[(size_t)k * N + n];
}

// combined su|sv weight: B[k][n] = n<128 ? w1[k][n] : w1[1280+k][n-128]; K=1280,N=256
__global__ void swizzle_w1(const float* __restrict__ w1, __bf16* __restrict__ out) {
    int idx = blockIdx.x * blockDim.x + threadIdx.x;
    if (idx >= TD_ * 256) return;
    int e = idx & 15, lane = (idx >> 4) & 31, frag = idx >> 9;
    int kt = frag % 40, nt = frag / 40;
    int n = nt * 16 + (lane & 15);
    int k = kt * 32 + apat(lane >= 16, e);
    float v = (n < 128) ? w1[(size_t)k * 128 + n]
                        : w1[(size_t)(TD_ + k) * 128 + (n - 128)];
    out[idx] = (__bf16)v;
}

// A-fragment swizzle of M^T where M is [128][128] row-major (batched): A[v][u]=M[u][v]
__global__ void swizzle_AT(const float* __restrict__ M, __bf16* __restrict__ out, int batch) {
    int idx = blockIdx.x * blockDim.x + threadIdx.x;
    if (idx >= batch * 16384) return;
    int bidx = idx >> 14, j = idx & 16383;
    int e = j & 15, lane = (j >> 4) & 31, frag = j >> 9;
    int kt = frag & 3, mt = frag >> 2;
    bool hi = lane >= 16;
    int v = mt * 16 + (lane & 15);
    int u = kt * 32 + apat(hi, e);
    out[idx] = (__bf16)M[(size_t)bidx * 16384 + u * 128 + v];
}

// h [BC][1280] f32 -> bf16 row-major
__global__ void h_to_bf(const float* __restrict__ h, __bf16* __restrict__ hbf) {
    int idx = blockIdx.x * blockDim.x + threadIdx.x;
    if (idx < BC_ * TD_) hbf[idx] = (__bf16)h[idx];
}

// h -> B-fragments per batch: frag = kt*80+nt (K-dim = u, col = (t,d))
__global__ void h_to_bfrag(const float* __restrict__ h, __bf16* __restrict__ out) {
    int idx = blockIdx.x * blockDim.x + threadIdx.x;
    if (idx >= B_ * 4 * 80 * 512) return;
    int b = idx / 163840, j = idx % 163840;
    int e = j & 15, lane = (j >> 4) & 31, frag = j >> 9;
    int nt = frag % 80, kt = frag / 80;
    int u = kt * 32 + apat(lane >= 16, e);
    int col = nt * 16 + (lane & 15);
    out[idx] = (__bf16)h[((size_t)(b * C_ + u)) * TD_ + col];
}

// ============ K1: RevIN stats
__global__ void revin_stats(const float* __restrict__ x, float* __restrict__ mean,
                            float* __restrict__ stdv) {
    int i = blockIdx.x * blockDim.x + threadIdx.x;
    if (i >= BC_) return;
    int b = i / C_, c = i % C_;
    float s = 0.f;
    for (int t = 0; t < T_; t++) s += x[((b * T_ + t) * C_ + c) * F_];
    float mu = s / T_;
    float v = 0.f;
    for (int t = 0; t < T_; t++) { float q = x[((b * T_ + t) * C_ + c) * F_] - mu; v += q * q; }
    mean[i] = mu;
    stdv[i] = sqrtf(v / T_ + EPS_);
}

// ============ K2: embedding + PE; h layout [B][C][T][D]
__global__ void embed_pe(const float* __restrict__ x, const float* __restrict__ mean,
                         const float* __restrict__ stdv, const float* __restrict__ rw,
                         const float* __restrict__ rb, const float* __restrict__ ew,
                         const float* __restrict__ eb, float* __restrict__ h) {
    int idx = blockIdx.x * blockDim.x + threadIdx.x;
    if (idx >= BC_ * TD_) return;
    int d = idx & (D_ - 1);
    int t = (idx / D_) % T_;
    int c = (idx / TD_) % C_;
    int b = idx / (TD_ * C_);
    float acc = eb[d];
    for (int f = 0; f < F_; f++) {
        float xv = x[((b * T_ + t) * C_ + c) * F_ + f];
        if (f == 0) { int i = b * C_ + c; xv = (xv - mean[i]) / stdv[i] * rw[c] + rb[c]; }
        acc += xv * ew[f * D_ + d];
    }
    h[idx] = acc + pe_val(t, d);
}

// ============ K3/K8: WMMA transformer block (T padded 20->32, zero pad rows)
__global__ void __launch_bounds__(128) xformer_block_wmma(
    float* __restrict__ hio,
    const __bf16* __restrict__ qwf, const float* __restrict__ qb,
    const __bf16* __restrict__ kwf, const float* __restrict__ kb,
    const __bf16* __restrict__ vwf, const float* __restrict__ vb,
    const float* __restrict__ g1, const float* __restrict__ b1,
    const float* __restrict__ g2, const float* __restrict__ b2,
    const __bf16* __restrict__ f1wf, const float* __restrict__ f1b,
    const __bf16* __restrict__ f2wf, const float* __restrict__ f2b) {
    __shared__ float xs[T_][D_];
    __shared__ __align__(16) __bf16 xbf[32][64];
    __shared__ __align__(16) __bf16 qld[32][64];
    __shared__ __align__(16) __bf16 kld[32][64];
    __shared__ __align__(16) __bf16 vT[64][32];
    __shared__ float sc[32][32];
    __shared__ __align__(16) __bf16 att[32][32];
    __shared__ __align__(16) __bf16 xn[32][64];
    __shared__ __align__(16) __bf16 h1[32][256];
    __shared__ float mu_[T_], rs_[T_];
    int tid = threadIdx.x, wave = tid >> 5, lane = tid & 31, m = lane & 15;
    bool hi = lane >= 16;
    float* base = hio + (size_t)blockIdx.x * TD_;

    for (int i = tid; i < 32 * 64; i += 128) {
        int r = i >> 6, d = i & 63;
        float v = (r < T_) ? base[r * 64 + d] : 0.f;
        if (r < T_) xs[r][d] = v;
        xbf[r][d] = (__bf16)v;
    }
    __syncthreads();
    // QKV: 3 mats x 2 mtiles x 4 ntiles; branch-free epilogue (ptr+index selects)
    for (int tile = wave; tile < 24; tile += 4) {
        int mat = tile >> 3, rem = tile & 7, mt = rem >> 2, nt = rem & 3;
        const __bf16* wf = (mat == 0) ? qwf : (mat == 1) ? kwf : vwf;
        const float* bb  = (mat == 0) ? qb  : (mat == 1) ? kb  : vb;
        int n = nt * 16 + m;
        float bv = bb[n];
        v8f acc; for (int i = 0; i < 8; i++) acc[i] = bv;
        for (int kt = 0; kt < 2; kt++) {
            v16bf A  = load_A_frag(&xbf[0][0], 64, mt * 16, kt * 32, lane);
            v16bf Bf = g_frag(wf, nt * 2 + kt, lane);
            acc = WMMA_BF16(A, Bf, acc);
        }
        __bf16* dst = (mat == 0) ? &qld[0][0] : (mat == 1) ? &kld[0][0] : &vT[0][0];
        bool isv = (mat == 2);
        for (int i = 0; i < 8; i++) {
            int row = mt * 16 + (hi ? 8 + i : i);
            int addr = isv ? (n * 32 + row) : (row * 64 + n);  // vT stored transposed
            dst[addr] = (__bf16)acc[i];
        }
    }
    __syncthreads();
    // scores = q @ k^T / 8 : k row-major is exactly Bt layout
    {
        int mt = wave >> 1, nt = wave & 1;
        v8f acc; for (int i = 0; i < 8; i++) acc[i] = 0.f;
        for (int kt = 0; kt < 2; kt++) {
            v16bf A  = load_A_frag(&qld[0][0], 64, mt * 16, kt * 32, lane);
            v16bf Bf = load_Bt_frag(&kld[0][0], 64, nt * 16, kt * 32, lane);
            acc = WMMA_BF16(A, Bf, acc);
        }
        for (int i = 0; i < 8; i++)
            sc[mt * 16 + (hi ? 8 + i : i)][nt * 16 + m] = acc[i] * 0.125f;
    }
    __syncthreads();
    // softmax rows (valid region), zero pad
    if (tid < T_) {
        float mx = -1e30f;
        for (int s = 0; s < T_; s++) mx = fmaxf(mx, sc[tid][s]);
        float sm = 0.f, ex[T_];
        for (int s = 0; s < T_; s++) { ex[s] = expf(sc[tid][s] - mx); sm += ex[s]; }
        float inv = 1.f / sm;
        for (int s = 0; s < T_; s++) att[tid][s] = (__bf16)(ex[s] * inv);
        for (int s = T_; s < 32; s++) att[tid][s] = (__bf16)0.f;
    } else if (tid < 32) {
        for (int s = 0; s < 32; s++) att[tid][s] = (__bf16)0.f;
    }
    __syncthreads();
    // attn @ v (+ residual)
    for (int tile = wave; tile < 8; tile += 4) {
        int mt = tile >> 2, nt = tile & 3;
        v8f acc; for (int i = 0; i < 8; i++) acc[i] = 0.f;
        v16bf A  = load_A_frag(&att[0][0], 32, mt * 16, 0, lane);
        v16bf Bf = load_Bt_frag(&vT[0][0], 32, nt * 16, 0, lane);
        acc = WMMA_BF16(A, Bf, acc);
        for (int i = 0; i < 8; i++) {
            int row = mt * 16 + (hi ? 8 + i : i);
            if (row < T_) xs[row][nt * 16 + m] += acc[i];
        }
    }
    __syncthreads();
    // LN1
    if (tid < T_) {
        float s = 0.f;
        for (int d = 0; d < D_; d++) s += xs[tid][d];
        float mm2 = s / D_, v = 0.f;
        for (int d = 0; d < D_; d++) { float q = xs[tid][d] - mm2; v += q * q; }
        mu_[tid] = mm2; rs_[tid] = rsqrtf(v / D_ + EPS_);
    }
    __syncthreads();
    for (int i = tid; i < 32 * 64; i += 128) {
        int r = i >> 6, d = i & 63;
        if (r < T_) {
            float v = (xs[r][d] - mu_[r]) * rs_[r] * g1[d] + b1[d];
            xs[r][d] = v;
            xn[r][d] = (__bf16)v;
        } else xn[r][d] = (__bf16)0.f;
    }
    __syncthreads();
    // FF1: 2 mt x 16 nt
    for (int tile = wave; tile < 32; tile += 4) {
        int mt = tile >> 4, nt = tile & 15;
        int n = nt * 16 + m;
        float bv = f1b[n];
        v8f acc; for (int i = 0; i < 8; i++) acc[i] = bv;
        for (int kt = 0; kt < 2; kt++) {
            v16bf A  = load_A_frag(&xn[0][0], 64, mt * 16, kt * 32, lane);
            v16bf Bf = g_frag(f1wf, nt * 2 + kt, lane);
            acc = WMMA_BF16(A, Bf, acc);
        }
        for (int i = 0; i < 8; i++)
            h1[mt * 16 + (hi ? 8 + i : i)][n] = (__bf16)fmaxf(acc[i], 0.f);
    }
    __syncthreads();
    // FF2: 2 mt x 4 nt, K=256 (+ residual)
    for (int tile = wave; tile < 8; tile += 4) {
        int mt = tile >> 2, nt = tile & 3;
        int n = nt * 16 + m;
        float bv = f2b[n];
        v8f acc; for (int i = 0; i < 8; i++) acc[i] = bv;
        for (int kt = 0; kt < 8; kt++) {
            v16bf A  = load_A_frag(&h1[0][0], 256, mt * 16, kt * 32, lane);
            v16bf Bf = g_frag(f2wf, nt * 8 + kt, lane);
            acc = WMMA_BF16(A, Bf, acc);
        }
        for (int i = 0; i < 8; i++) {
            int row = mt * 16 + (hi ? 8 + i : i);
            if (row < T_) xs[row][n] += acc[i];
        }
    }
    __syncthreads();
    // LN2 + store
    if (tid < T_) {
        float s = 0.f;
        for (int d = 0; d < D_; d++) s += xs[tid][d];
        float mm2 = s / D_, v = 0.f;
        for (int d = 0; d < D_; d++) { float q = xs[tid][d] - mm2; v += q * q; }
        mu_[tid] = mm2; rs_[tid] = rsqrtf(v / D_ + EPS_);
    }
    __syncthreads();
    for (int i = tid; i < TD_; i += 128) {
        int t = i >> 6, d = i & 63;
        base[i] = (xs[t][d] - mu_[t]) * rs_[t] * g2[d] + b2[d];
    }
}

// ============ K4: su/sv via WMMA GEMM [2048x1280]@[1280x256]
__global__ void __launch_bounds__(256) adaptor_proj_wmma(
    const __bf16* __restrict__ hbf, const __bf16* __restrict__ w1f,
    float* __restrict__ su, float* __restrict__ sv) {
    int wave = threadIdx.x >> 5, lane = threadIdx.x & 31, m = lane & 15;
    bool hi = lane >= 16;
    int gw = blockIdx.x * 8 + wave;   // 256 waves
    int mt = gw >> 1;                 // 0..127
    int ng = gw & 1;                  // 0 -> su cols, 1 -> sv cols
    v8f acc[8];
    for (int nt = 0; nt < 8; nt++)
        for (int i = 0; i < 8; i++) acc[nt][i] = 0.f;
    const __bf16* arow = hbf + (size_t)mt * 16 * TD_;
    for (int kt = 0; kt < 40; kt++) {
        v16bf A = load_A_frag(arow, TD_, 0, kt * 32, lane);
        for (int nt = 0; nt < 8; nt++) {
            v16bf Bf = g_frag(w1f, (ng * 8 + nt) * 40 + kt, lane);
            acc[nt] = WMMA_BF16(A, Bf, acc[nt]);
        }
    }
    float* out = ng ? sv : su;
    for (int nt = 0; nt < 8; nt++)
        for (int i = 0; i < 8; i++) {
            int row = mt * 16 + (hi ? 8 + i : i);
            out[(size_t)row * 128 + nt * 16 + m] = acc[nt][i];
        }
}

// ============ K5: pairwise MLP 128->256->64->1 with bf16 WMMA
__global__ void __launch_bounds__(256) pairwise_mlp(
    const float* __restrict__ su, const float* __restrict__ sv,
    const float* __restrict__ b1v, const __bf16* __restrict__ w2f,
    const float* __restrict__ b2v, const __bf16* __restrict__ w3f,
    const float* __restrict__ b3v, const float* __restrict__ w4,
    const float* __restrict__ b4v, const float* __restrict__ A_a,
    float* __restrict__ wedge) {
    __shared__ __align__(16) __bf16 lds_t[8][16][32];
    int wave = threadIdx.x >> 5, lane = threadIdx.x & 31;
    int b = blockIdx.x >> 7, u = blockIdx.x & 127;
    int v0 = wave * 16, m = lane & 15;
    bool hi = lane >= 16;
    const float* suR = su + ((size_t)(b * C_ + u)) * 128;
    const float* svR = sv + ((size_t)(b * C_ + (v0 + m))) * 128;

    // layer-1 activation in A layout; chunks of 8 K are contiguous -> float4 loads
    v16bf a1[4];
    for (int kt = 0; kt < 4; kt++) {
        int k0 = kt * 32;
        for (int c = 0; c < 2; c++) {
            int cs = k0 + c * 16 + (hi ? 8 : 0);
            float4 s0 = *(const float4*)(suR + cs), s1 = *(const float4*)(suR + cs + 4);
            float4 v0q = *(const float4*)(svR + cs), v1q = *(const float4*)(svR + cs + 4);
            float4 b0 = *(const float4*)(b1v + cs), b1q = *(const float4*)(b1v + cs + 4);
            int eb = c * 8;
            a1[kt][eb + 0] = (__bf16)fmaxf(s0.x + v0q.x + b0.x, 0.f);
            a1[kt][eb + 1] = (__bf16)fmaxf(s0.y + v0q.y + b0.y, 0.f);
            a1[kt][eb + 2] = (__bf16)fmaxf(s0.z + v0q.z + b0.z, 0.f);
            a1[kt][eb + 3] = (__bf16)fmaxf(s0.w + v0q.w + b0.w, 0.f);
            a1[kt][eb + 4] = (__bf16)fmaxf(s1.x + v1q.x + b1q.x, 0.f);
            a1[kt][eb + 5] = (__bf16)fmaxf(s1.y + v1q.y + b1q.y, 0.f);
            a1[kt][eb + 6] = (__bf16)fmaxf(s1.z + v1q.z + b1q.z, 0.f);
            a1[kt][eb + 7] = (__bf16)fmaxf(s1.w + v1q.w + b1q.w, 0.f);
        }
    }
    v8f acc3[4];
    for (int nt = 0; nt < 4; nt++) {
        float bb = b3v[nt * 16 + m];
        for (int i = 0; i < 8; i++) acc3[nt][i] = bb;
    }
    for (int kt3 = 0; kt3 < 8; kt3++) {
        float bb0 = b2v[(2 * kt3) * 16 + m];
        float bb1 = b2v[(2 * kt3 + 1) * 16 + m];
        v8f ac0, ac1;
        for (int i = 0; i < 8; i++) { ac0[i] = bb0; ac1[i] = bb1; }
        for (int kt = 0; kt < 4; kt++) {
            v16bf f0 = g_frag(w2f, (2 * kt3) * 4 + kt, lane);
            v16bf f1 = g_frag(w2f, (2 * kt3 + 1) * 4 + kt, lane);
            ac0 = WMMA_BF16(a1[kt], f0, ac0);
            ac1 = WMMA_BF16(a1[kt], f1, ac1);
        }
        for (int i = 0; i < 8; i++) {
            int mm = hi ? 8 + i : i;
            lds_t[wave][mm][m]      = (__bf16)fmaxf(ac0[i], 0.f);
            lds_t[wave][mm][16 + m] = (__bf16)fmaxf(ac1[i], 0.f);
        }
        asm volatile("s_wait_dscnt 0x0" ::: "memory");
        FragU a3;
        int c0 = hi ? 8 : 0;
        a3.u[0] = *(const uint4*)&lds_t[wave][m][c0];
        a3.u[1] = *(const uint4*)&lds_t[wave][m][c0 + 16];
        for (int nt = 0; nt < 4; nt++) {
            v16bf w3u = g_frag(w3f, nt * 8 + kt3, lane);
            acc3[nt] = WMMA_BF16(a3.v, w3u, acc3[nt]);
        }
    }
    float b4s = b4v[0];
    for (int i = 0; i < 8; i++) {
        float p = 0.f;
        for (int nt = 0; nt < 4; nt++) p += fmaxf(acc3[nt][i], 0.f) * w4[nt * 16 + m];
        p += __shfl_xor(p, 1); p += __shfl_xor(p, 2);
        p += __shfl_xor(p, 4); p += __shfl_xor(p, 8);
        if (m == 0) {
            int mm = hi ? 8 + i : i;
            int v = v0 + mm;
            float s = 1.f / (1.f + expf(-(p + b4s)));
            wedge[((size_t)(b * C_ + u)) * C_ + v] = s * A_a[u * C_ + v];
        }
    }
}

// ============ K6a: spatial aggregation GEMMs (wedge^T@h, A_m^T@h), beta-folded bf16 out
__global__ void __launch_bounds__(256) spatial_agg_wmma(
    const __bf16* __restrict__ hfrag, const __bf16* __restrict__ wTf,
    const __bf16* __restrict__ amTf, const float* __restrict__ hf32,
    const float* __restrict__ beta2, const float* __restrict__ beta3,
    __bf16* __restrict__ agg_a, __bf16* __restrict__ agg_mh) {
    int wave = threadIdx.x >> 5, lane = threadIdx.x & 31, m = lane & 15;
    bool hi = lane >= 16;
    int b = blockIdx.x >> 3, mt = blockIdx.x & 7;
    v16bf Aw[4], Am[4];
    for (int kt = 0; kt < 4; kt++) {
        Aw[kt] = g_frag(wTf + (size_t)b * 16384, mt * 4 + kt, lane);
        Am[kt] = g_frag(amTf, mt * 4 + kt, lane);
    }
    float b2s = beta2[0], b3s = beta3[0];
    for (int nt = wave; nt < 80; nt += 8) {
        v8f aa, am;
        for (int i = 0; i < 8; i++) { aa[i] = 0.f; am[i] = 0.f; }
        for (int kt = 0; kt < 4; kt++) {
            v16bf Bf = g_frag(hfrag + (size_t)b * 163840, kt * 80 + nt, lane);
            aa = WMMA_BF16(Aw[kt], Bf, aa);
            am = WMMA_BF16(Am[kt], Bf, am);
        }
        for (int i = 0; i < 8; i++) {
            int v = mt * 16 + (hi ? 8 + i : i);
            int col = nt * 16 + m;
            size_t g = ((size_t)(b * C_ + v)) * TD_ + col;
            agg_a[g]  = (__bf16)(b2s * aa[i]);
            agg_mh[g] = (__bf16)(b3s * am[i] * hf32[g]);
        }
    }
}

// ============ K6b: fc_add/fc_mod mix [40960x64]@[64x64]x2 + beta1*h + PE -> z
__global__ void __launch_bounds__(256) fc_mix_wmma(
    const __bf16* __restrict__ agg_a, const __bf16* __restrict__ agg_mh,
    const __bf16* __restrict__ fawf, const __bf16* __restrict__ fmwf,
    const float* __restrict__ fab, const float* __restrict__ fmb,
    const float* __restrict__ beta1, const float* __restrict__ beta2,
    const float* __restrict__ beta3, const float* __restrict__ hf32,
    float* __restrict__ z) {
    int wave = threadIdx.x >> 5, lane = threadIdx.x & 31, m = lane & 15;
    bool hi = lane >= 16;
    int r0 = (blockIdx.x * 8 + wave) * 16;
    v16bf Aa[2], Am[2];
    for (int kt = 0; kt < 2; kt++) {
        Aa[kt] = load_A_frag(agg_a  + (size_t)r0 * 64, 64, 0, kt * 32, lane);
        Am[kt] = load_A_frag(agg_mh + (size_t)r0 * 64, 64, 0, kt * 32, lane);
    }
    float b1s = beta1[0], b2s = beta2[0], b3s = beta3[0];
    for (int nt = 0; nt < 4; nt++) {
        int n = nt * 16 + m;
        float bv = b2s * fab[n] + b3s * fmb[n];
        v8f acc; for (int i = 0; i < 8; i++) acc[i] = bv;
        for (int kt = 0; kt < 2; kt++) {
            acc = WMMA_BF16(Aa[kt], g_frag(fawf, nt * 2 + kt, lane), acc);
            acc = WMMA_BF16(Am[kt], g_frag(fmwf, nt * 2 + kt, lane), acc);
        }
        for (int i = 0; i < 8; i++) {
            int row = r0 + (hi ? 8 + i : i);
            int t = row % T_;
            size_t g = (size_t)row * 64 + n;
            z[g] = acc[i] + b1s * hf32[g] + pe_val(t, n);
        }
    }
}

// ============ K9: readout + RevIN denorm
__global__ void readout(const float* __restrict__ r, const float* __restrict__ ow,
                        const float* __restrict__ ob, const float* __restrict__ rw,
                        const float* __restrict__ rb, const float* __restrict__ mean,
                        const float* __restrict__ stdv, float* __restrict__ out) {
    int idx = blockIdx.x * blockDim.x + threadIdx.x;
    if (idx >= B_ * T_ * C_) return;
    int c = idx & 127;
    int t = (idx >> 7) % T_;
    int b = idx / (C_ * T_);
    const float* row = r + (((size_t)(b * C_ + c)) * T_ + t) * D_;
    float a = ob[0];
    for (int d = 0; d < D_; d++) a += row[d] * ow[d];
    int i = b * C_ + c;
    out[idx] = (a - rb[c]) / rw[c] * stdv[i] + mean[i];
}

extern "C" void kernel_launch(void* const* d_in, const int* in_sizes, int n_in,
                              void* d_out, int out_size, void* d_ws, size_t ws_size,
                              hipStream_t stream) {
    const float* x        = (const float*)d_in[0];
    const float* revin_w  = (const float*)d_in[1];
    const float* revin_b  = (const float*)d_in[2];
    const float* te_ew    = (const float*)d_in[3];
    const float* te_eb    = (const float*)d_in[4];
    const float* A_a      = (const float*)d_in[33];
    const float* A_m      = (const float*)d_in[34];
    const float* fc_add_w = (const float*)d_in[35];
    const float* fc_add_b = (const float*)d_in[36];
    const float* fc_mod_w = (const float*)d_in[37];
    const float* fc_mod_b = (const float*)d_in[38];
    const float* beta1    = (const float*)d_in[39];
    const float* beta2    = (const float*)d_in[40];
    const float* beta3    = (const float*)d_in[41];
    const float* ad_w1    = (const float*)d_in[42];
    const float* ad_b1    = (const float*)d_in[43];
    const float* ad_w2    = (const float*)d_in[44];
    const float* ad_b2    = (const float*)d_in[45];
    const float* ad_w3    = (const float*)d_in[46];
    const float* ad_b3    = (const float*)d_in[47];
    const float* ad_w4    = (const float*)d_in[48];
    const float* ad_b4    = (const float*)d_in[49];
    const float* out_w    = (const float*)d_in[50];
    const float* out_b    = (const float*)d_in[51];
    float* out = (float*)d_out;

    char* ws = (char*)d_ws;
    size_t off = 0;
    auto carve = [&](size_t bytes) -> void* {
        void* p = ws + off;
        off = (off + bytes + 255) & ~(size_t)255;
        return p;
    };
    float* mean  = (float*)carve(BC_ * 4);
    float* stdv  = (float*)carve(BC_ * 4);
    float* h     = (float*)carve((size_t)BC_ * TD_ * 4);
    float* z     = (float*)carve((size_t)BC_ * TD_ * 4);
    float* su    = (float*)carve((size_t)BC_ * 128 * 4);
    float* sv    = (float*)carve((size_t)BC_ * 128 * 4);
    float* wedge = (float*)carve((size_t)BC_ * C_ * 4);
    __bf16* w2f  = (__bf16*)carve(128 * 256 * 2);
    __bf16* w3f  = (__bf16*)carve(256 * 64 * 2);
    __bf16* teqf = (__bf16*)carve(64 * 64 * 2);
    __bf16* tekf = (__bf16*)carve(64 * 64 * 2);
    __bf16* tevf = (__bf16*)carve(64 * 64 * 2);
    __bf16* tef1 = (__bf16*)carve(64 * 256 * 2);
    __bf16* tef2 = (__bf16*)carve(256 * 64 * 2);
    __bf16* trqf = (__bf16*)carve(64 * 64 * 2);
    __bf16* trkf = (__bf16*)carve(64 * 64 * 2);
    __bf16* trvf = (__bf16*)carve(64 * 64 * 2);
    __bf16* trf1 = (__bf16*)carve(64 * 256 * 2);
    __bf16* trf2 = (__bf16*)carve(256 * 64 * 2);
    __bf16* faf  = (__bf16*)carve(64 * 64 * 2);
    __bf16* fmf  = (__bf16*)carve(64 * 64 * 2);
    __bf16* w1f  = (__bf16*)carve((size_t)TD_ * 256 * 2);
    __bf16* hbf  = (__bf16*)carve((size_t)BC_ * TD_ * 2);
    __bf16* hfrag= (__bf16*)carve((size_t)BC_ * TD_ * 2);
    __bf16* amf  = (__bf16*)carve(16384 * 2);
    __bf16* wTf  = (__bf16*)carve((size_t)B_ * 16384 * 2);
    __bf16* agga = (__bf16*)carve((size_t)ROWS_ * 64 * 2);
    __bf16* aggm = (__bf16*)carve((size_t)ROWS_ * 64 * 2);
    (void)ws_size; (void)in_sizes; (void)n_in; (void)out_size;

    auto swz = [&](const float* W, __bf16* o, int K, int N) {
        swizzle_w<<<(K * N + 255) / 256, 256, 0, stream>>>(W, o, K, N);
    };
    // weight swizzles
    swz(ad_w2, w2f, 128, 256);
    swz(ad_w3, w3f, 256, 64);
    swz((const float*)d_in[5],  teqf, 64, 64);
    swz((const float*)d_in[7],  tekf, 64, 64);
    swz((const float*)d_in[9],  tevf, 64, 64);
    swz((const float*)d_in[15], tef1, 64, 256);
    swz((const float*)d_in[17], tef2, 256, 64);
    swz((const float*)d_in[19], trqf, 64, 64);
    swz((const float*)d_in[21], trkf, 64, 64);
    swz((const float*)d_in[23], trvf, 64, 64);
    swz((const float*)d_in[29], trf1, 64, 256);
    swz((const float*)d_in[31], trf2, 256, 64);
    swz(fc_add_w, faf, 64, 64);
    swz(fc_mod_w, fmf, 64, 64);
    swizzle_w1<<<(TD_ * 256 + 255) / 256, 256, 0, stream>>>(ad_w1, w1f);
    swizzle_AT<<<(16384 + 255) / 256, 256, 0, stream>>>(A_m, amf, 1);

    // pipeline
    revin_stats<<<(BC_ + 255) / 256, 256, 0, stream>>>(x, mean, stdv);
    embed_pe<<<(BC_ * TD_ + 255) / 256, 256, 0, stream>>>(
        x, mean, stdv, revin_w, revin_b, te_ew, te_eb, h);
    xformer_block_wmma<<<BC_, 128, 0, stream>>>(
        h, teqf, (const float*)d_in[6], tekf, (const float*)d_in[8],
        tevf, (const float*)d_in[10],
        (const float*)d_in[11], (const float*)d_in[12],
        (const float*)d_in[13], (const float*)d_in[14],
        tef1, (const float*)d_in[16], tef2, (const float*)d_in[18]);
    h_to_bf<<<(BC_ * TD_ + 255) / 256, 256, 0, stream>>>(h, hbf);
    h_to_bfrag<<<(B_ * 163840 + 255) / 256, 256, 0, stream>>>(h, hfrag);
    adaptor_proj_wmma<<<32, 256, 0, stream>>>(hbf, w1f, su, sv);
    pairwise_mlp<<<BC_, 256, 0, stream>>>(
        su, sv, ad_b1, w2f, ad_b2, w3f, ad_b3, ad_w4, ad_b4, A_a, wedge);
    swizzle_AT<<<(B_ * 16384 + 255) / 256, 256, 0, stream>>>(wedge, wTf, B_);
    spatial_agg_wmma<<<B_ * 8, 256, 0, stream>>>(
        hfrag, wTf, amf, h, beta2, beta3, agga, aggm);
    fc_mix_wmma<<<ROWS_ / 128, 256, 0, stream>>>(
        agga, aggm, faf, fmf, fc_add_b, fc_mod_b, beta1, beta2, beta3, h, z);
    xformer_block_wmma<<<BC_, 128, 0, stream>>>(
        z, trqf, (const float*)d_in[20], trkf, (const float*)d_in[22],
        trvf, (const float*)d_in[24],
        (const float*)d_in[25], (const float*)d_in[26],
        (const float*)d_in[27], (const float*)d_in[28],
        trf1, (const float*)d_in[30], trf2, (const float*)d_in[32]);
    readout<<<(B_ * T_ * C_ + 255) / 256, 256, 0, stream>>>(
        z, out_w, out_b, revin_w, revin_b, mean, stdv, out);
}
